// Attention_74680891343387
// MI455X (gfx1250) — matrix operand order repaired and checked by reading, not compiled, not run on hardware
//
#include <hip/hip_runtime.h>
#include <hip/hip_bf16.h>
#include <math.h>

typedef __attribute__((ext_vector_type(16))) _Float16 v16h;
typedef __attribute__((ext_vector_type(8)))  float    v8f;

#define B_   4
#define C_   32
#define H_   96
#define W_   96
#define HW_  (H_*W_)      // 9216
#define L_   HW_
#define BL_  (B_*L_)      // 36864
#define DI_  64
#define NS_  16
#define OUT_ 64
#define CS_  288          // scan chunk length
#define NC_  32           // chunks per sequence (CS_*NC_ == L_)

// ---- gfx1250 async global->LDS staging (guarded; falls back to vector copy) --
#if defined(__has_builtin)
#  if __has_builtin(__builtin_amdgcn_global_load_async_to_lds_b128)
#    define HAVE_ASYNC_LDS 1
#  endif
#endif
#ifndef HAVE_ASYNC_LDS
#  define HAVE_ASYNC_LDS 0
#endif

#if HAVE_ASYNC_LDS
typedef int v4i_vs __attribute__((vector_size(16)));
typedef __attribute__((address_space(1))) v4i_vs* gptr128_t;   // "__device__" in diags
typedef __attribute__((address_space(3))) v4i_vs* lptr128_t;   // "__shared__" in diags
#endif

__device__ __forceinline__ void stage16(void* lds_dst, const void* gsrc, int n16,
                                        int tid, int nth)
{
#if HAVE_ASYNC_LDS
    for (int i = tid; i < n16; i += nth) {
        __builtin_amdgcn_global_load_async_to_lds_b128(
            (gptr128_t)((const char*)gsrc + (size_t)i * 16),
            (lptr128_t)((char*)lds_dst + (size_t)i * 16),
            0, 0);
    }
#  if defined(__has_builtin) && __has_builtin(__builtin_amdgcn_s_wait_asynccnt)
    __builtin_amdgcn_s_wait_asynccnt(0);
#  else
    asm volatile("s_wait_asynccnt 0" ::: "memory");
#  endif
#else
    const uint4* s = (const uint4*)gsrc;
    uint4*       d = (uint4*)lds_dst;
    for (int i = tid; i < n16; i += nth) d[i] = s[i];
#endif
    __syncthreads();
}

// ---------------------------------------------------------------------------
// 0) convert weights to f16, transposed [k][n]-major for WMMA B-fragments
// ---------------------------------------------------------------------------
__global__ void k_cvtw(const float* __restrict__ w_in, const float* __restrict__ w_out,
                       const float* __restrict__ w_conv,
                       _Float16* __restrict__ w_in_t,   // [32][128]
                       _Float16* __restrict__ w_out_t,  // [64][32]
                       _Float16* __restrict__ wc)       // [9][32][64]
{
    int i = blockIdx.x * blockDim.x + threadIdx.x;
    if (i < 32 * 128) {
        int k = i / 128, n = i % 128;
        w_in_t[i] = (_Float16)w_in[n * 32 + k];
    } else if (i < 32 * 128 + 64 * 32) {
        int j = i - 4096;
        int k = j / 32, n = j % 32;
        w_out_t[j] = (_Float16)w_out[n * 64 + k];
    } else if (i < 4096 + 2048 + 9 * 32 * 64) {
        int j = i - 6144;
        int tap = j / 2048, r = j % 2048, k = r / 64, n = r % 64;
        wc[j] = (_Float16)w_conv[n * 288 + k * 9 + tap];
    }
}

// ---------------------------------------------------------------------------
// 1) dilated depthwise 7x7 conv (dil=3,pad=9) + LayerNorm over C, emit f16
// ---------------------------------------------------------------------------
__global__ void k_dwconv_ln(const float* __restrict__ x, const float* __restrict__ w_att,
                            const float* __restrict__ b_att, const float* __restrict__ ln_w,
                            const float* __restrict__ ln_b, _Float16* __restrict__ xn_h)
{
    int wid  = (blockIdx.x * blockDim.x + threadIdx.x) >> 5;  // token
    int lane = threadIdx.x & 31;                               // channel
    int b = wid / HW_, hw = wid % HW_;
    int h = hw / W_, w = hw % W_;
    const float* xp = x + (size_t)(b * C_ + lane) * HW_;
    const float* wk = w_att + lane * 49;
    float acc = b_att[lane];
#pragma unroll
    for (int i = 0; i < 7; ++i) {
        int h2 = h + i * 3 - 9;
        if (h2 < 0 || h2 >= H_) continue;
#pragma unroll
        for (int j = 0; j < 7; ++j) {
            int w2 = w + j * 3 - 9;
            if (w2 < 0 || w2 >= W_) continue;
            acc += xp[h2 * W_ + w2] * wk[i * 7 + j];
        }
    }
    float s = acc, s2 = acc * acc;
#pragma unroll
    for (int m = 16; m >= 1; m >>= 1) { s += __shfl_xor(s, m); s2 += __shfl_xor(s2, m); }
    float mu  = s * (1.0f / 32.0f);
    float var = s2 * (1.0f / 32.0f) - mu * mu;
    float xn  = (acc - mu) * rsqrtf(var + 1e-5f) * ln_w[lane] + ln_b[lane];
    xn_h[(size_t)wid * 32 + lane] = (_Float16)xn;
}

// ---------------------------------------------------------------------------
// 2) in-projection xz = xn @ w_in.T : (BL,32)x(32,128), WMMA K=32 one shot
//    weights staged block-wide into LDS (async path on gfx1250 when available)
// ---------------------------------------------------------------------------
__global__ void k_inproj(const _Float16* __restrict__ xn_h, const _Float16* __restrict__ w_t,
                         float* __restrict__ xz)
{
    __shared__ _Float16 lds_w[32 * 128];                     // 8 KB
    stage16(lds_w, w_t, 32 * 128 * 2 / 16, threadIdx.x, blockDim.x);

    int wave = (blockIdx.x * blockDim.x + threadIdx.x) >> 5;
    int lane = threadIdx.x & 31;
    int mTile = wave >> 3, nTile = wave & 7;
    int m = lane & 15, hi = lane >> 4;
    int token = mTile * 16 + m;
    const _Float16* arow = xn_h + (size_t)token * 32 + (hi ? 8 : 0);
    v16h a, bf;
#pragma unroll
    for (int i = 0; i < 8; ++i) { a[i] = arow[i]; a[8 + i] = arow[16 + i]; }
    int n  = nTile * 16 + m;
    int k0 = hi * 8;
#pragma unroll
    for (int i = 0; i < 16; ++i) bf[i] = lds_w[(k0 + i + (i >= 8 ? 8 : 0)) * 128 + n];
    v8f c = {};
    c = __builtin_amdgcn_wmma_f32_16x16x32_f16(false, a, false, bf, (short)0, c, false, false);
#pragma unroll
    for (int r = 0; r < 8; ++r) {
        int mm = r + hi * 8;
        xz[(size_t)(mTile * 16 + mm) * 128 + n] = c[r];
    }
}

// ---------------------------------------------------------------------------
// 3) per-direction: (anti)causal depthwise conv1d + silu, x-proj, softplus
// ---------------------------------------------------------------------------
__global__ void k_dirprep(const float* __restrict__ xz, const float* __restrict__ cw,
                          const float* __restrict__ cb, const float* __restrict__ wx,
                          const float* __restrict__ wdt, const float* __restrict__ bdt,
                          float* __restrict__ xc_o, float* __restrict__ dl_o,
                          float* __restrict__ Bo, float* __restrict__ Co, int bwd)
{
    int t = blockIdx.x * blockDim.x + threadIdx.x;
    if (t >= BL_) return;
    int b = t / L_, l = t % L_;
    float xc[DI_];
#pragma unroll
    for (int d = 0; d < DI_; ++d) xc[d] = cb[d];
#pragma unroll
    for (int k = 0; k < 4; ++k) {
        int l2 = bwd ? (l + 3 - k) : (l + k - 3);
        if (l2 < 0 || l2 >= L_) continue;
        const float* row = xz + (size_t)(b * L_ + l2) * 128;
        for (int d = 0; d < DI_; ++d) xc[d] += row[d] * cw[d * 4 + k];
    }
#pragma unroll
    for (int d = 0; d < DI_; ++d) { float v = xc[d]; xc[d] = v / (1.0f + __expf(-v)); }
    float dbc[34];
    for (int j = 0; j < 34; ++j) {
        float acc = 0.0f;
        const float* wr = wx + j * DI_;
        for (int d = 0; d < DI_; ++d) acc += xc[d] * wr[d];
        dbc[j] = acc;
    }
    float* xo  = xc_o + (size_t)t * DI_;
    float* dlo = dl_o + (size_t)t * DI_;
    for (int d = 0; d < DI_; ++d) {
        xo[d] = xc[d];
        float u = dbc[0] * wdt[d * 2] + dbc[1] * wdt[d * 2 + 1] + bdt[d];
        dlo[d] = (u > 20.0f) ? u : __logf(1.0f + __expf(u));
    }
    float* Bp = Bo + (size_t)t * NS_;
    float* Cp = Co + (size_t)t * NS_;
    for (int n = 0; n < NS_; ++n) { Bp[n] = dbc[2 + n]; Cp[n] = dbc[18 + n]; }
}

// ---------------------------------------------------------------------------
// 4a) chunked scan phase A: per-chunk local end state + cumulative decay P
// ---------------------------------------------------------------------------
__global__ void k_scan_part(const float* __restrict__ xc, const float* __restrict__ dl,
                            const float* __restrict__ Bc, const float* __restrict__ A_log,
                            float* __restrict__ hend, float* __restrict__ pend, int bwd)
{
    int wave = (blockIdx.x * blockDim.x + threadIdx.x) >> 5;
    int lane = threadIdx.x & 31;
    int half = lane >> 4, n = lane & 15;
    int task = wave * 2 + half;               // 0..8191
    int seq = task >> 5, chunk = task & (NC_ - 1);
    int b = seq >> 6, d = seq & 63;
    float Av = -__expf(A_log[d * NS_ + n]);
    float h = 0.0f, P = 1.0f;
    const size_t base = (size_t)b * L_;
    int c0 = chunk * CS_;
    for (int i = 0; i < CS_; ++i) {
        int t = bwd ? (c0 + CS_ - 1 - i) : (c0 + i);
        size_t tok = base + t;
        float dlt = dl[tok * DI_ + d];
        float xv  = xc[tok * DI_ + d];
        float Bv  = Bc[tok * NS_ + n];
        float dA  = __expf(dlt * Av);
        h = dA * h + dlt * Bv * xv;
        P *= dA;
    }
    hend[(size_t)task * NS_ + n] = h;
    pend[(size_t)task * NS_ + n] = P;
}

// ---------------------------------------------------------------------------
// 4b) carry scan over chunk summaries: elementwise per (seq, state) lane
// ---------------------------------------------------------------------------
__global__ void k_scan_carry(const float* __restrict__ hend, const float* __restrict__ pend,
                             float* __restrict__ hin, int bwd)
{
    int tid = blockIdx.x * blockDim.x + threadIdx.x;      // 4096 lanes
    int seq = tid >> 4, n = tid & 15;
    float h = 0.0f;
#pragma unroll
    for (int i = 0; i < NC_; ++i) {
        int c = bwd ? (NC_ - 1 - i) : i;
        size_t idx = ((size_t)seq * NC_ + c) * NS_ + n;
        hin[idx] = h;
        h = pend[idx] * h + hend[idx];
    }
}

// ---------------------------------------------------------------------------
// 4c) chunked scan phase C: re-scan with correct carry-in, produce y
// ---------------------------------------------------------------------------
__global__ void k_scan_fix(const float* __restrict__ xc, const float* __restrict__ dl,
                           const float* __restrict__ Bc, const float* __restrict__ Cc,
                           const float* __restrict__ A_log, const float* __restrict__ Dp,
                           const float* __restrict__ hin, float* __restrict__ y, int bwd)
{
    int wave = (blockIdx.x * blockDim.x + threadIdx.x) >> 5;
    int lane = threadIdx.x & 31;
    int half = lane >> 4, n = lane & 15;
    int task = wave * 2 + half;
    int seq = task >> 5, chunk = task & (NC_ - 1);
    int b = seq >> 6, d = seq & 63;
    float Av = -__expf(A_log[d * NS_ + n]);
    float Dv = Dp[d];
    float h  = hin[(size_t)task * NS_ + n];
    const size_t base = (size_t)b * L_;
    int c0 = chunk * CS_;
    for (int i = 0; i < CS_; ++i) {
        int t = bwd ? (c0 + CS_ - 1 - i) : (c0 + i);
        size_t tok = base + t;
        // stream-ahead prefetch (clamped; speculative -> safe)
        int tp = bwd ? (t - 32 < 0 ? 0 : t - 32) : (t + 32 >= L_ ? L_ - 1 : t + 32);
        size_t tokp = base + tp;
        __builtin_prefetch(&dl[tokp * DI_ + d], 0, 0);
        __builtin_prefetch(&xc[tokp * DI_ + d], 0, 0);
        float dlt = dl[tok * DI_ + d];
        float xv  = xc[tok * DI_ + d];
        float Bv  = Bc[tok * NS_ + n];
        float Cv  = Cc[tok * NS_ + n];
        h = __expf(dlt * Av) * h + dlt * Bv * xv;
        float p = h * Cv;
        p += __shfl_xor(p, 1); p += __shfl_xor(p, 2);
        p += __shfl_xor(p, 4); p += __shfl_xor(p, 8);
        if (n == 0) y[tok * DI_ + d] = p + xv * Dv;
    }
}

// ---------------------------------------------------------------------------
// 5) out-projection: att2 = ((y_f+y_b)*silu(z)) @ w_out.T, WMMA K=64 (2 steps)
// ---------------------------------------------------------------------------
__global__ void k_outproj(const float* __restrict__ yf, const float* __restrict__ yb,
                          const float* __restrict__ xz, const _Float16* __restrict__ wo_t,
                          float* __restrict__ att2)
{
    int wave = (blockIdx.x * blockDim.x + threadIdx.x) >> 5;
    int lane = threadIdx.x & 31;
    int mTile = wave >> 1, nTile = wave & 1;
    int m = lane & 15, hi = lane >> 4;
    int token = mTile * 16 + m;
    size_t tb = (size_t)token;
    v8f c = {};
#pragma unroll
    for (int kk = 0; kk < 2; ++kk) {
        int kbase = kk * 32 + (hi ? 8 : 0);
        v16h a, bf;
#pragma unroll
        for (int i = 0; i < 8; ++i) {
            int k1 = kbase + i, k2 = kbase + 16 + i;
            float z1 = xz[tb * 128 + 64 + k1];
            float z2 = xz[tb * 128 + 64 + k2];
            float u1 = (yf[tb * DI_ + k1] + yb[tb * DI_ + k1]) * (z1 / (1.0f + __expf(-z1)));
            float u2 = (yf[tb * DI_ + k2] + yb[tb * DI_ + k2]) * (z2 / (1.0f + __expf(-z2)));
            a[i] = (_Float16)u1;  a[8 + i] = (_Float16)u2;
        }
        int n  = nTile * 16 + m;
        int k0 = kk * 32 + hi * 8;
#pragma unroll
        for (int i = 0; i < 16; ++i) bf[i] = wo_t[(size_t)(k0 + i + (i >= 8 ? 8 : 0)) * 32 + n];
        c = __builtin_amdgcn_wmma_f32_16x16x32_f16(false, a, false, bf, (short)0, c, false, false);
    }
    int n = nTile * 16 + m;
#pragma unroll
    for (int r = 0; r < 8; ++r) {
        int mm = r + hi * 8;
        att2[(size_t)(mTile * 16 + mm) * 32 + n] = c[r];
    }
}

// ---------------------------------------------------------------------------
// 6) channel mean/max pool over concat(att1=x, att2)
// ---------------------------------------------------------------------------
__global__ void k_pool(const float* __restrict__ x, const float* __restrict__ att2,
                       float* __restrict__ pooled)
{
    int t = blockIdx.x * blockDim.x + threadIdx.x;
    if (t >= BL_) return;
    int b = t / HW_, hw = t % HW_;
    float sm = 0.0f, mx = -3.4e38f;
    for (int c = 0; c < C_; ++c) {
        float v1 = x[(size_t)(b * C_ + c) * HW_ + hw];
        float v2 = att2[(size_t)(b * HW_ + hw) * 32 + c];
        sm += v1 + v2;
        mx = fmaxf(mx, fmaxf(v1, v2));
    }
    pooled[(size_t)(b * 2 + 0) * HW_ + hw] = sm * (1.0f / 64.0f);
    pooled[(size_t)(b * 2 + 1) * HW_ + hw] = mx;
}

// ---------------------------------------------------------------------------
// 7) SE 7x7 conv + sigmoid + blend -> f16 token-major feed for final conv
// ---------------------------------------------------------------------------
__global__ void k_se(const float* __restrict__ x, const float* __restrict__ att2,
                     const float* __restrict__ pooled, const float* __restrict__ w_se,
                     const float* __restrict__ b_se, _Float16* __restrict__ outp_h)
{
    int t = blockIdx.x * blockDim.x + threadIdx.x;
    if (t >= BL_) return;
    int b = t / HW_, hw = t % HW_, h = hw / W_, w = hw % W_;
    float s0 = b_se[0], s1 = b_se[1];
    for (int ic = 0; ic < 2; ++ic) {
        const float* pp = pooled + (size_t)(b * 2 + ic) * HW_;
        for (int i = 0; i < 7; ++i) {
            int h2 = h + i - 3;
            if (h2 < 0 || h2 >= H_) continue;
            for (int j = 0; j < 7; ++j) {
                int w2 = w + j - 3;
                if (w2 < 0 || w2 >= W_) continue;
                float pv = pp[h2 * W_ + w2];
                s0 += pv * w_se[(0 * 2 + ic) * 49 + i * 7 + j];
                s1 += pv * w_se[(1 * 2 + ic) * 49 + i * 7 + j];
            }
        }
    }
    s0 = 1.0f / (1.0f + __expf(-s0));
    s1 = 1.0f / (1.0f + __expf(-s1));
    _Float16* orow = outp_h + (size_t)(b * HW_ + hw) * 32;
    for (int c = 0; c < C_; ++c) {
        float v = x[(size_t)(b * C_ + c) * HW_ + hw] * s0
                + att2[(size_t)(b * HW_ + hw) * 32 + c] * s1;
        orow[c] = (_Float16)v;
    }
}

// ---------------------------------------------------------------------------
// 8) final 3x3 conv 32->64 as implicit GEMM (K=288 = 9 taps x K32 WMMA)
// ---------------------------------------------------------------------------
__global__ void k_fconv(const _Float16* __restrict__ outp_h, const _Float16* __restrict__ wc_h,
                        const float* __restrict__ b_conv, float* __restrict__ out)
{
    __shared__ _Float16 lds_w[9 * 32 * 64];                  // 36 KB
    stage16(lds_w, wc_h, 9 * 32 * 64 * 2 / 16, threadIdx.x, blockDim.x);

    int wave = (blockIdx.x * blockDim.x + threadIdx.x) >> 5;
    int lane = threadIdx.x & 31;
    int mTile = wave >> 2, nTile = wave & 3;
    int m = lane & 15, hi = lane >> 4;
    int token = mTile * 16 + m;
    int b = token / HW_, hw = token % HW_, h = hw / W_, w = hw % W_;
    int kb = hi ? 8 : 0;
    v8f c = {};
#pragma unroll
    for (int ky = 0; ky < 3; ++ky) {
#pragma unroll
        for (int kx = 0; kx < 3; ++kx) {
            int h2 = h + ky - 1, w2 = w + kx - 1;
            bool ok = (h2 >= 0) && (h2 < H_) && (w2 >= 0) && (w2 < W_);
            int h2c = min(max(h2, 0), H_ - 1), w2c = min(max(w2, 0), W_ - 1);
            const _Float16* row = outp_h + (size_t)(b * HW_ + h2c * W_ + w2c) * 32;
            v16h a, bf;
#pragma unroll
            for (int i = 0; i < 8; ++i) {
                a[i]     = ok ? row[kb + i]      : (_Float16)0.0f;
                a[8 + i] = ok ? row[kb + 16 + i] : (_Float16)0.0f;
            }
            const _Float16* wt = lds_w + (ky * 3 + kx) * 32 * 64;
            int n  = nTile * 16 + m;
            int k0 = hi * 8;
#pragma unroll
            for (int i = 0; i < 16; ++i) bf[i] = wt[(k0 + i + (i >= 8 ? 8 : 0)) * 64 + n];
            c = __builtin_amdgcn_wmma_f32_16x16x32_f16(false, a, false, bf, (short)0, c, false, false);
        }
    }
    int oc = nTile * 16 + m;
    float bias = b_conv[oc];
#pragma unroll
    for (int r = 0; r < 8; ++r) {
        int mm   = r + hi * 8;
        int tok2 = mTile * 16 + mm;
        int bb = tok2 / HW_, hw2 = tok2 % HW_;
        out[(size_t)(bb * OUT_ + oc) * HW_ + hw2] = c[r] + bias;
    }
}

// ---------------------------------------------------------------------------
extern "C" void kernel_launch(void* const* d_in, const int* in_sizes, int n_in,
                              void* d_out, int out_size, void* d_ws, size_t ws_size,
                              hipStream_t stream)
{
    const float* x       = (const float*)d_in[0];
    const float* w_att   = (const float*)d_in[1];
    const float* b_att   = (const float*)d_in[2];
    const float* ln_w    = (const float*)d_in[3];
    const float* ln_b    = (const float*)d_in[4];
    const float* w_in    = (const float*)d_in[5];
    const float* cw_f    = (const float*)d_in[6];
    const float* cb_f    = (const float*)d_in[7];
    const float* wx_f    = (const float*)d_in[8];
    const float* wdt_f   = (const float*)d_in[9];
    const float* bdt_f   = (const float*)d_in[10];
    const float* Alog_f  = (const float*)d_in[11];
    const float* D_f     = (const float*)d_in[12];
    const float* cw_b    = (const float*)d_in[13];
    const float* cb_b    = (const float*)d_in[14];
    const float* wx_b    = (const float*)d_in[15];
    const float* wdt_b   = (const float*)d_in[16];
    const float* bdt_b   = (const float*)d_in[17];
    const float* Alog_b  = (const float*)d_in[18];
    const float* D_b     = (const float*)d_in[19];
    const float* w_out   = (const float*)d_in[20];
    const float* w_se    = (const float*)d_in[21];
    const float* b_se    = (const float*)d_in[22];
    const float* w_conv  = (const float*)d_in[23];
    const float* b_conv  = (const float*)d_in[24];
    float* out = (float*)d_out;

    // workspace bump allocator (~98 MB, L2-resident on MI455X's 192 MB GL2)
    char* p = (char*)d_ws;
    auto alloc = [&](size_t bytes) { char* r = p; p += (bytes + 255) & ~(size_t)255; return (void*)r; };
    _Float16* xn_h    = (_Float16*)alloc((size_t)BL_ * 32 * 2);
    float*    xz      = (float*)   alloc((size_t)BL_ * 128 * 4);
    float*    xc_f    = (float*)   alloc((size_t)BL_ * DI_ * 4);
    float*    dl_f    = (float*)   alloc((size_t)BL_ * DI_ * 4);
    float*    Bc_f    = (float*)   alloc((size_t)BL_ * NS_ * 4);
    float*    Cc_f    = (float*)   alloc((size_t)BL_ * NS_ * 4);
    float*    xc_b    = (float*)   alloc((size_t)BL_ * DI_ * 4);
    float*    dl_b    = (float*)   alloc((size_t)BL_ * DI_ * 4);
    float*    Bc_b    = (float*)   alloc((size_t)BL_ * NS_ * 4);
    float*    Cc_b    = (float*)   alloc((size_t)BL_ * NS_ * 4);
    float*    y_f     = (float*)   alloc((size_t)BL_ * DI_ * 4);
    float*    y_b     = (float*)   alloc((size_t)BL_ * DI_ * 4);
    float*    att2    = (float*)   alloc((size_t)BL_ * 32 * 4);
    float*    pooled  = (float*)   alloc((size_t)B_ * 2 * HW_ * 4);
    _Float16* outp_h  = (_Float16*)alloc((size_t)BL_ * 32 * 2);
    _Float16* w_in_t  = (_Float16*)alloc((size_t)32 * 128 * 2);
    _Float16* w_out_t = (_Float16*)alloc((size_t)64 * 32 * 2);
    _Float16* wc_h    = (_Float16*)alloc((size_t)9 * 32 * 64 * 2);
    float*    hend_f  = (float*)   alloc((size_t)256 * NC_ * NS_ * 4);
    float*    pend_f  = (float*)   alloc((size_t)256 * NC_ * NS_ * 4);
    float*    hin_f   = (float*)   alloc((size_t)256 * NC_ * NS_ * 4);
    float*    hend_b  = (float*)   alloc((size_t)256 * NC_ * NS_ * 4);
    float*    pend_b  = (float*)   alloc((size_t)256 * NC_ * NS_ * 4);
    float*    hin_b   = (float*)   alloc((size_t)256 * NC_ * NS_ * 4);

    // 0) weight conversion
    k_cvtw<<<96, 256, 0, stream>>>(w_in, w_out, w_conv, w_in_t, w_out_t, wc_h);

    // 1) dwconv + LN: one wave per token
    k_dwconv_ln<<<BL_ / 8, 256, 0, stream>>>(x, w_att, b_att, ln_w, ln_b, xn_h);

    // 2) in-projection: 2304 m-tiles x 8 n-tiles = 18432 waves
    k_inproj<<<BL_ / 16, 256, 0, stream>>>(xn_h, w_in_t, xz);

    // 3) direction prep
    k_dirprep<<<BL_ / 256, 256, 0, stream>>>(xz, cw_f, cb_f, wx_f, wdt_f, bdt_f,
                                             xc_f, dl_f, Bc_f, Cc_f, 0);
    k_dirprep<<<BL_ / 256, 256, 0, stream>>>(xz, cw_b, cb_b, wx_b, wdt_b, bdt_b,
                                             xc_b, dl_b, Bc_b, Cc_b, 1);

    // 4) chunked parallel scans: 8192 half-wave tasks (4096 waves) per phase
    k_scan_part <<<512, 256, 0, stream>>>(xc_f, dl_f, Bc_f, Alog_f, hend_f, pend_f, 0);
    k_scan_part <<<512, 256, 0, stream>>>(xc_b, dl_b, Bc_b, Alog_b, hend_b, pend_b, 1);
    k_scan_carry<<<16, 256, 0, stream>>>(hend_f, pend_f, hin_f, 0);
    k_scan_carry<<<16, 256, 0, stream>>>(hend_b, pend_b, hin_b, 1);
    k_scan_fix  <<<512, 256, 0, stream>>>(xc_f, dl_f, Bc_f, Cc_f, Alog_f, D_f, hin_f, y_f, 0);
    k_scan_fix  <<<512, 256, 0, stream>>>(xc_b, dl_b, Bc_b, Cc_b, Alog_b, D_b, hin_b, y_b, 1);

    // 5) out-projection: 4608 waves
    k_outproj<<<576, 256, 0, stream>>>(y_f, y_b, xz, w_out_t, att2);

    // 6) pool, 7) SE+blend
    k_pool<<<BL_ / 256, 256, 0, stream>>>(x, att2, pooled);
    k_se<<<BL_ / 256, 256, 0, stream>>>(x, att2, pooled, w_se, b_se, outp_h);

    // 8) final 3x3 conv: 9216 waves
    k_fconv<<<1152, 256, 0, stream>>>(outp_h, wc_h, b_conv, out);
}
